// NaiveSSEMultiHeadAttention_17566416241402
// MI455X (gfx1250) — compile-verified, matrix operand order checked by reading, not run on hardware
//
#include <hip/hip_runtime.h>

// ---------------------------------------------------------------------------
// Problem constants (from reference): B=4 S=4096 DM=1024 H=16 P=32 R=8 D=64 k=4
// ---------------------------------------------------------------------------
#define BBATCH 4
#define SS   4096
#define DMM  1024
#define HH   16
#define PP   32
#define RR   8
#define DD   64
#define KSEL 4
#define TOK  (BBATCH * SS)      // 16384 tokens

typedef __attribute__((ext_vector_type(16))) __bf16 bf16x16;
typedef __attribute__((ext_vector_type(8)))  float  v8f;

union FragBF16 {                // 16 bf16 = 32 bytes = two uint4 chunks
    bf16x16 v;
    uint4   q[2];
};

static __device__ __forceinline__ unsigned short f2bf(float f) {
    unsigned int u = __float_as_uint(f);
    u += 0x7fffu + ((u >> 16) & 1u);   // round-to-nearest-even
    return (unsigned short)(u >> 16);
}

// ---------------------------------------------------------------------------
// CDNA5 async global->LDS B128 copy via inline asm (GLOBAL_LOAD_ASYNC_TO_LDS,
// ASYNCcnt-tracked; VDST = 32-bit LDS byte offset, VADDR = 64-bit global addr,
// SADDR disabled -> "off").  Generic LDS pointers truncate to the DS offset
// per the ISA aperture rule.
// ---------------------------------------------------------------------------
static __device__ __forceinline__ void async_copy_b128(void* lds, const void* gbl) {
    unsigned int       loff = (unsigned int)(unsigned long long)lds;
    unsigned long long ga   = (unsigned long long)gbl;
    asm volatile("global_load_async_to_lds_b128 %0, %1, off"
                 :: "v"(loff), "v"(ga) : "memory");
}
static __device__ __forceinline__ void wait_async_zero() {
    asm volatile("s_wait_asynccnt 0x0" ::: "memory");
}

// ---------------------------------------------------------------------------
// Kernel 0: W_out (fp32, [K=1024][N=1024]) -> Wt (bf16, transposed [N][K])
// so that the GEMM B-operand has contiguous K per n-row (fragment-friendly).
// ---------------------------------------------------------------------------
__global__ __launch_bounds__(256) void convert_wout(
    const float* __restrict__ W, unsigned short* __restrict__ Wt)
{
    int idx = blockIdx.x * 256 + threadIdx.x;    // 0 .. 1024*1024-1
    int k = idx >> 10;
    int n = idx & 1023;
    Wt[((size_t)n << 10) + k] = f2bf(W[idx]);    // coalesced read
}

// ---------------------------------------------------------------------------
// Kernel 1: fused router + top-4 gate + sparse expert attention.
// One wave32 per (token, head). lane == expert index p for the router stage.
// Only the 4 gated experts are scored/combined (8x FLOP cut vs dense).
// Heads written as bf16 into workspace for the WMMA GEMM.
// ---------------------------------------------------------------------------
__global__ __launch_bounds__(256) void fused_router_attn(
    const float* __restrict__ x,        // [TOK][DM]
    const float* __restrict__ Kst,      // [H][P][R][D]
    const float* __restrict__ Vst,      // [H][P][R][D]
    const float* __restrict__ Wr,       // [H][D][P]
    unsigned short* __restrict__ headsBf) // [TOK][DM] bf16
{
    __shared__ float xs_all[8][DD];
    const int lane = threadIdx.x & 31;
    const int wid  = threadIdx.x >> 5;
    const long long gw = (long long)blockIdx.x * 8 + wid;  // wave over TOK*H
    const int t = (int)(gw >> 4);   // token
    const int h = (int)(gw & 15);   // head

    // stage xh into LDS (broadcast source for the dot products)
    const float* xh = x + (size_t)t * DMM + (size_t)h * DD;
    float* xs = xs_all[wid];
    xs[lane]      = xh[lane];
    xs[lane + 32] = xh[lane + 32];
    __syncthreads();

    // ---- router logits: lane p owns logit_p -------------------------------
    const float* wr = Wr + (size_t)h * DD * PP;
    float logit = 0.0f;
    #pragma unroll 8
    for (int d = 0; d < DD; ++d)
        logit = fmaf(xs[d], wr[d * PP + lane], logit);

    // ---- top-4 threshold via 4 rounds of wave-max -------------------------
    float cur = logit;
    float thresh = 0.0f, gmax = 0.0f;
    for (int i = 0; i < KSEL; ++i) {
        float m = cur;
        for (int off = 16; off; off >>= 1)
            m = fmaxf(m, __shfl_xor(m, off, 32));
        if (i == 0) gmax = m;
        thresh = m;
        unsigned long long bal = __ballot(cur == m);
        int first = __ffsll(bal) - 1;
        if (lane == first) cur = -__builtin_inff();
    }

    // ---- gate softmax over kept experts -----------------------------------
    bool keep = (logit >= thresh);
    float e = keep ? __expf(logit - gmax) : 0.0f;
    float esum = e;
    for (int off = 16; off; off >>= 1) esum += __shfl_xor(esum, off, 32);
    float gate = e / esum;
    unsigned int mask = (unsigned int)__ballot(keep);

    // ---- sparse expert loop: scores, softmax over R, combine V ------------
    float acc0 = 0.0f, acc1 = 0.0f;
    const int r   = lane >> 2;       // 8 slots, 4 lanes each
    const int seg = lane & 3;        // 16 d-elements per lane
    const float scale = 0.125f;      // 1/sqrt(64)

    while (mask) {
        const int p = __ffs(mask) - 1;
        mask &= mask - 1;

        const size_t base = (((size_t)h * PP + p) * RR) * DD;
        // score_r = xh . K[h][p][r][:]  (lanes split d into 4 segments)
        const float* kr = Kst + base + (size_t)r * DD + seg * 16;
        float s = 0.0f;
        #pragma unroll
        for (int i = 0; i < 16; ++i)
            s = fmaf(xs[seg * 16 + i], kr[i], s);
        s += __shfl_xor(s, 1, 32);
        s += __shfl_xor(s, 2, 32);
        s *= scale;

        // softmax over r (each r value replicated on 4 lanes)
        float ms = s;
        for (int off = 16; off; off >>= 1)
            ms = fmaxf(ms, __shfl_xor(ms, off, 32));
        float ee = __expf(s - ms);
        float se = ee;
        for (int off = 16; off; off >>= 1) se += __shfl_xor(se, off, 32);
        se *= 0.25f;                               // each r counted 4x
        const float gp = __shfl(gate, p, 32);
        const float w  = gp * ee / se;

        // combine: head[d] += sum_r w_r * V[h][p][r][d]
        const float* vp = Vst + base;
        #pragma unroll
        for (int rr = 0; rr < RR; ++rr) {
            const float wr8 = __shfl(w, rr * 4, 32);
            acc0 = fmaf(wr8, vp[rr * DD + lane],      acc0);
            acc1 = fmaf(wr8, vp[rr * DD + lane + 32], acc1);
        }
    }

    const size_t ob = (size_t)t * DMM + (size_t)h * DD;
    headsBf[ob + lane]      = f2bf(acc0);
    headsBf[ob + lane + 32] = f2bf(acc1);
}

// ---------------------------------------------------------------------------
// Kernel 2: C[16384,1024] = A(bf16)[16384,1024] x Wt(bf16)[1024(n),1024(k)]^T
//           + bias, fp32 out.  v_wmma_f32_16x16x32_bf16, 128x128 block tile,
//           8 waves, each wave a 32x64 patch (2x4 WMMA tiles, K-loop of 32).
//           Tile fills use GLOBAL_LOAD_ASYNC_TO_LDS_B128 (no VGPR staging).
// ---------------------------------------------------------------------------
#define BN  128
#define BK  32
#define LDK (BK + 8)            // pad LDS rows to 80B: conflict-free b128 reads

__global__ __launch_bounds__(256) void gemm_wmma_bf16(
    const unsigned short* __restrict__ A,   // [16384][1024] bf16
    const unsigned short* __restrict__ Bt,  // [1024][1024]  bf16  (Wt[n][k])
    const float* __restrict__ bias,         // [1024]
    float* __restrict__ C)                  // [16384][1024] fp32
{
    __shared__ __align__(16) unsigned short As[128][LDK]; // 128x40 ushort=10KB
    __shared__ __align__(16) unsigned short Bs[128][LDK]; // 10KB

    const int tid  = threadIdx.x;
    const int lane = tid & 31;
    const int wid  = tid >> 5;
    const int wm   = wid & 3;           // 4 M-slices of 32 rows
    const int wn   = wid >> 2;          // 2 N-slices of 64 cols
    const int m0   = blockIdx.x * 128;
    const int n0   = blockIdx.y * BN;
    const int l16  = lane & 15;
    const int klo  = (lane < 16) ? 0 : 8;   // fragment K-chunk select (ushorts)

    v8f acc[2][4] = {};

    for (int k0 = 0; k0 < DMM; k0 += BK) {
        // ---- tile fill: 512 x 16B async DMA chunks each for A and B -------
        #pragma unroll
        for (int i = 0; i < 2; ++i) {
            const int c   = tid + i * 256;
            const int row = c >> 2;
            const int sg  = c & 3;
            async_copy_b128(&As[row][sg * 8],
                            A  + (((size_t)(m0 + row)) << 10) + k0 + sg * 8);
            async_copy_b128(&Bs[row][sg * 8],
                            Bt + (((size_t)(n0 + row)) << 10) + k0 + sg * 8);
        }
        // prefetch next K-tile into L2 (global_prefetch_b8)
        if (k0 + BK < DMM) {
            __builtin_prefetch(A  + (((size_t)(m0 + (tid >> 1))) << 10) + k0 + BK, 0, 1);
            __builtin_prefetch(Bt + (((size_t)(n0 + (tid >> 1))) << 10) + k0 + BK, 0, 1);
        }
        wait_async_zero();          // drain this thread's async copies
        __syncthreads();            // make tile visible block-wide

        // ---- build fragments (two ds_load_b128 per fragment per lane) -----
        FragBF16 a[2], b[4];
        #pragma unroll
        for (int i = 0; i < 2; ++i) {
            const unsigned short* src = &As[wm * 32 + i * 16 + l16][0];
            a[i].q[0] = *(const uint4*)(src + klo);        // K 0..7 / 8..15
            a[i].q[1] = *(const uint4*)(src + klo + 16);   // K 16..23 / 24..31
        }
        #pragma unroll
        for (int j = 0; j < 4; ++j) {
            const unsigned short* src = &Bs[wn * 64 + j * 16 + l16][0];
            b[j].q[0] = *(const uint4*)(src + klo);
            b[j].q[1] = *(const uint4*)(src + klo + 16);
        }

        // ---- 8 WMMA ops per wave per K-step -------------------------------
        #pragma unroll
        for (int i = 0; i < 2; ++i)
            #pragma unroll
            for (int j = 0; j < 4; ++j)
                acc[i][j] = __builtin_amdgcn_wmma_f32_16x16x32_bf16(
                    false, a[i].v, false, b[j].v,
                    (short)0, acc[i][j], false, false);
        __syncthreads();
    }

    // ---- epilogue: C/D layout -> lane<16: M=v, lane>=16: M=v+8 -------------
    const int mbase = m0 + wm * 32;
    const int nbase = n0 + wn * 64;
    #pragma unroll
    for (int i = 0; i < 2; ++i) {
        #pragma unroll
        for (int j = 0; j < 4; ++j) {
            const int ncol = nbase + j * 16 + l16;
            const float bv = bias[ncol];
            #pragma unroll
            for (int v = 0; v < 8; ++v) {
                const int m = mbase + i * 16 + ((lane < 16) ? v : v + 8);
                C[((size_t)m << 10) + ncol] = acc[i][j][v] + bv;
            }
        }
    }
}

// ---------------------------------------------------------------------------
// Host launcher
// ---------------------------------------------------------------------------
extern "C" void kernel_launch(void* const* d_in, const int* in_sizes, int n_in,
                              void* d_out, int out_size, void* d_ws, size_t ws_size,
                              hipStream_t stream) {
    const float* x    = (const float*)d_in[0];
    const float* Kst  = (const float*)d_in[1];
    const float* Vst  = (const float*)d_in[2];
    const float* Wr   = (const float*)d_in[3];
    const float* Wout = (const float*)d_in[4];
    const float* bout = (const float*)d_in[5];
    float* out = (float*)d_out;

    // workspace: heads bf16 [16384*1024] then Wt bf16 [1024*1024]
    unsigned short* headsBf = (unsigned short*)d_ws;
    unsigned short* Wt = (unsigned short*)((char*)d_ws +
                         (size_t)TOK * DMM * sizeof(unsigned short));

    // K0: transpose+convert W_out
    convert_wout<<<(DMM * DMM) / 256, 256, 0, stream>>>(Wout, Wt);

    // K1: fused router + sparse attention (one wave per token-head)
    fused_router_attn<<<(TOK * HH) / 8, 256, 0, stream>>>(x, Kst, Vst, Wr, headsBf);

    // K2: WMMA bf16 GEMM + bias
    dim3 grid(TOK / 128, DMM / BN);
    gemm_wmma_bf16<<<grid, 256, 0, stream>>>(headsBf, Wt, bout, out);
}
